// RNNGenerator_38792144617753
// MI455X (gfx1250) — matrix operand order repaired and checked by reading, not compiled, not run on hardware
//
#include <hip/hip_runtime.h>
#include <hip/hip_bf16.h>
#include <math.h>

// ---------------------------------------------------------------------------
// Bidirectional 2-layer LSTM + dense head for MI455X (gfx1250), bf16 WMMA.
// All GEMM work uses v_wmma_f32_16x16x32_bf16 (fp32 accumulate). K loops are
// per-segment, branch-free, ping-pong double-buffered: loads for k+64/k+96
// overlap the WMMAs for k/k+32.
// ---------------------------------------------------------------------------

typedef __bf16 bf16_t;
typedef __attribute__((ext_vector_type(16))) __bf16 v16bf;
typedef __attribute__((ext_vector_type(8)))  __bf16 v8bf;
typedef __attribute__((ext_vector_type(8)))  float  v8f;

constexpr int cB = 128, cT = 512, cU = 512, cV = 32, cE = 256;

// ---- WMMA helpers ---------------------------------------------------------

__device__ __forceinline__ v8f wmma_bf16(v16bf a, v16bf b, v8f c) {
  // 8 args: (neg_a, A, neg_b, B, c_mod, C, reuse_a, reuse_b)
  return __builtin_amdgcn_wmma_f32_16x16x32_bf16(
      false, a, false, b, (short)0, c, false, false);
}

// A tile: 16x32 bf16, row-major source, `A` points at tile origin.
// ISA layout: lanes 0-15 row=lane, K {0..7, 16..23};
//             lanes 16-31 row=lane-16, K {8..15, 24..31}.
__device__ __forceinline__ v16bf load_a_tile(const bf16_t* __restrict__ A,
                                             int lda, int lane) {
  int l = lane & 15, hi = lane >> 4;
  const bf16_t* p = A + l * lda + hi * 8;
  v8bf lo = *(const v8bf*)p;
  v8bf hh = *(const v8bf*)(p + 16);
  v16bf r;
#pragma unroll
  for (int j = 0; j < 8; ++j) { r[j] = lo[j]; r[8 + j] = hh[j]; }
  return r;
}

// B tile: 32x16 bf16 from a pre-TRANSPOSED weight Wt[N][K] (row-major, ld=K).
// ISA layout: lanes 0-15 => N=lane, K {0..7, 16..23}; lanes 16-31 => N=lane-16, K {8..15, 24..31}.
__device__ __forceinline__ v16bf load_b_tile(const bf16_t* __restrict__ Bt,
                                             int ldb, int lane) {
  int n = lane & 15, hi = lane >> 4;
  const bf16_t* p = Bt + n * ldb + hi * 8;
  v8bf x0 = *(const v8bf*)p;
  v8bf x1 = *(const v8bf*)(p + 16);
  v16bf r;
#pragma unroll
  for (int j = 0; j < 8; ++j) { r[j] = x0[j]; r[8 + j] = x1[j]; }
  return r;
}

// One K-segment of a GEMM with G column tiles sharing the A operand.
// `len` is a multiple of 64 and >= 64; steady-state loop has unconditional
// prefetch and the epilogue just drains the two buffers (branch-free).
template <int G>
__device__ __forceinline__ void gemm_seg(v8f (&acc)[G],
                                         const bf16_t* __restrict__ src, int lda,
                                         const bf16_t* const (&wg)[G], int ldw,
                                         int kw, int len, int lane) {
  v16bf aP = load_a_tile(src, lda, lane);
  v16bf aQ = load_a_tile(src + 32, lda, lane);
  v16bf bP[G], bQ[G];
#pragma unroll
  for (int g = 0; g < G; ++g) {
    bP[g] = load_b_tile(wg[g] + kw, ldw, lane);
    bQ[g] = load_b_tile(wg[g] + kw + 32, ldw, lane);
  }
  int k = 0;
  for (; k + 64 < len; k += 64) {
    // consume P @ k, refill P <- k+64 (overlaps Q's WMMAs)
#pragma unroll
    for (int g = 0; g < G; ++g) acc[g] = wmma_bf16(aP, bP[g], acc[g]);
    aP = load_a_tile(src + k + 64, lda, lane);
#pragma unroll
    for (int g = 0; g < G; ++g) bP[g] = load_b_tile(wg[g] + kw + k + 64, ldw, lane);
    // consume Q @ k+32, refill Q <- k+96 (k+96 <= len-32 here)
#pragma unroll
    for (int g = 0; g < G; ++g) acc[g] = wmma_bf16(aQ, bQ[g], acc[g]);
    aQ = load_a_tile(src + k + 96, lda, lane);
#pragma unroll
    for (int g = 0; g < G; ++g) bQ[g] = load_b_tile(wg[g] + kw + k + 96, ldw, lane);
  }
  // drain
#pragma unroll
  for (int g = 0; g < G; ++g) acc[g] = wmma_bf16(aP, bP[g], acc[g]);
#pragma unroll
  for (int g = 0; g < G; ++g) acc[g] = wmma_bf16(aQ, bQ[g], acc[g]);
}

__device__ __forceinline__ float sigmoidf_(float x) {
  return 1.0f / (1.0f + __expf(-x));
}
__device__ __forceinline__ float tanhf_(float x) {
  return 1.0f - 2.0f / (__expf(2.0f * x) + 1.0f);
}

// ---- prep kernels ---------------------------------------------------------

// x0[t][b][e] = bf16(emb[x[b][t]][e] * scale)
__global__ void embed_kernel(const int* __restrict__ x,
                             const float* __restrict__ emb,
                             bf16_t* __restrict__ x0, float scale) {
  unsigned i = blockIdx.x * blockDim.x + threadIdx.x;  // < T*B*E = 2^24
  int e = (int)(i & (cE - 1));
  unsigned tb = i >> 8;            // cE = 256
  int b = (int)(tb & (cB - 1));
  int t = (int)(tb >> 7);          // cB = 128
  int tok = x[b * cT + t];
  x0[i] = (bf16_t)(emb[tok * cE + e] * scale);
}

// Combined transposed LSTM weight: dst[4U][Kin+U] bf16;
// cols [0,Kin) from Ksrc[Kin][4U], cols [Kin,Kin+U) from Rsrc[U][4U].
__global__ void pack_lstm_w(const float* __restrict__ Ksrc,
                            const float* __restrict__ Rsrc,
                            bf16_t* __restrict__ dst, int Kin) {
  int Ktot = Kin + cU;
  unsigned i = blockIdx.x * blockDim.x + threadIdx.x;
  if (i >= (unsigned)(4 * cU * Ktot)) return;
  int k = (int)(i % Ktot);
  int n = (int)(i / Ktot);
  float v = (k < Kin) ? Ksrc[k * (4 * cU) + n]
                      : Rsrc[(k - Kin) * (4 * cU) + n];
  dst[i] = (bf16_t)v;
}

// dst[N][K] bf16 = transpose of W[K][N] fp32
__global__ void pack_dense_w(const float* __restrict__ W,
                             bf16_t* __restrict__ dst, int K, int N) {
  unsigned i = blockIdx.x * blockDim.x + threadIdx.x;
  if (i >= (unsigned)(K * N)) return;
  int k = (int)(i % K);
  int n = (int)(i / K);
  dst[i] = (bf16_t)W[k * N + n];
}

// ---- fused LSTM step (both directions) ------------------------------------
// One wave owns a (16-batch x 16-unit) tile for one direction and computes the
// 4 gate tiles z = [inA | inB | h_prev] @ Wt^T + bias, then the cell update.
__global__ void __launch_bounds__(256)
lstm_step(const bf16_t* __restrict__ inA,   // [T][B][KA]
          const bf16_t* __restrict__ inB,   // [T][B][KB] or nullptr
          int KA, int KB,
          const bf16_t* __restrict__ Wt_f,  // [4U][KA+KB+U] transposed bf16
          const bf16_t* __restrict__ Wt_b,
          const float* __restrict__ bias_f, // [4U]
          const float* __restrict__ bias_b,
          bf16_t* __restrict__ hseq_f,      // [T][B][U]
          bf16_t* __restrict__ hseq_b,
          float* __restrict__ c_f,          // [B][U] fp32, in-place
          float* __restrict__ c_b,
          int t) {
  int wave = (int)((blockIdx.x * blockDim.x + threadIdx.x) >> 5);
  int lane = (int)(threadIdx.x & 31);
  constexpr int colTiles = cU / 16;                 // 32
  constexpr int wavesPerDir = colTiles * (cB / 16); // 256
  int dir = wave / wavesPerDir;
  if (dir > 1) return;
  int w = wave % wavesPerDir;
  int ct = w % colTiles, rt = w / colTiles;

  const bf16_t* Wt   = dir ? Wt_b : Wt_f;
  const float*  bias = dir ? bias_b : bias_f;
  bf16_t*       hseq = dir ? hseq_b : hseq_f;
  float*        cbuf = dir ? c_b : c_f;
  int tt    = dir ? (cT - 1 - t) : t;      // actual time index
  int tprev = dir ? (tt + 1) : (tt - 1);
  int Ktot  = KA + KB + cU;
  int row0 = rt * 16, col0 = ct * 16;
  int n = lane & 15;

  // init accumulators with per-column gate bias
  v8f acc[4];
#pragma unroll
  for (int g = 0; g < 4; ++g) {
    float bv = bias[g * cU + col0 + n];
#pragma unroll
    for (int r = 0; r < 8; ++r) acc[g][r] = bv;
  }

  // per-gate weight row bases
  const bf16_t* wg[4];
#pragma unroll
  for (int g = 0; g < 4; ++g) wg[g] = Wt + (size_t)(g * cU + col0) * Ktot;

  // K segments: input A, (optional) input B, recurrent h_prev
  gemm_seg<4>(acc, inA + (tt * cB + row0) * KA, KA, wg, Ktot, 0, KA, lane);
  if (inB != nullptr) {
    gemm_seg<4>(acc, inB + (tt * cB + row0) * KB, KB, wg, Ktot, KA, KB, lane);
  }
  if (t > 0) {
    int KB_eff = (inB != nullptr) ? KB : 0;
    gemm_seg<4>(acc, hseq + (tprev * cB + row0) * cU, cU, wg, Ktot,
                KA + KB_eff, cU, lane);
  }

  // cell update; element (r,lane): row = row0 + (lane>>4)*8 + r, col = col0 + n
  int mrow = row0 + ((lane >> 4) << 3);
#pragma unroll
  for (int r = 0; r < 8; ++r) {
    int brow = mrow + r;
    float ig = sigmoidf_(acc[0][r]);
    float fg = sigmoidf_(acc[1][r]);
    float gg = tanhf_(acc[2][r]);
    float og = sigmoidf_(acc[3][r]);
    float cp = (t == 0) ? 0.0f : cbuf[brow * cU + col0 + n];
    float cn = fg * cp + ig * gg;
    cbuf[brow * cU + col0 + n] = cn;
    float hn = og * tanhf_(cn);
    hseq[(tt * cB + brow) * cU + col0 + n] = (bf16_t)hn;
  }
}

// ---- dense head -----------------------------------------------------------
// out1 = relu(concat(hf,hb) @ W1t^T + d1), bf16 out, N=K=1024.
__global__ void __launch_bounds__(256)
dense1_kernel(const bf16_t* __restrict__ hf, const bf16_t* __restrict__ hb,
              const bf16_t* __restrict__ Wt, const float* __restrict__ bias,
              bf16_t* __restrict__ out) {
  constexpr int N = 1024, K = 1024;
  int wave = (int)((blockIdx.x * blockDim.x + threadIdx.x) >> 5);
  int lane = (int)(threadIdx.x & 31);
  constexpr int colGroups = N / 64;  // each wave: 1 row tile x 4 col tiles
  int rt = wave / colGroups;
  int cg = wave % colGroups;
  int row0 = rt * 16, col0 = cg * 64;
  int n = lane & 15;
  v8f acc[4];
#pragma unroll
  for (int j = 0; j < 4; ++j) {
    float bv = bias[col0 + j * 16 + n];
#pragma unroll
    for (int r = 0; r < 8; ++r) acc[j][r] = bv;
  }
  const bf16_t* wg[4];
#pragma unroll
  for (int j = 0; j < 4; ++j) wg[j] = Wt + (size_t)(col0 + j * 16) * K;

  gemm_seg<4>(acc, hf + row0 * cU, cU, wg, K, 0, cU, lane);
  gemm_seg<4>(acc, hb + row0 * cU, cU, wg, K, cU, cU, lane);

  int mrow = row0 + ((lane >> 4) << 3);
#pragma unroll
  for (int j = 0; j < 4; ++j)
#pragma unroll
    for (int r = 0; r < 8; ++r) {
      float v = acc[j][r];
      v = v > 0.0f ? v : 0.0f;
      out[(size_t)(mrow + r) * N + col0 + j * 16 + n] = (bf16_t)v;
    }
}

// logits = out1 @ W2t^T + d2, fp32 out, written [B][T][V] from time-major rows.
__global__ void __launch_bounds__(256)
dense2_kernel(const bf16_t* __restrict__ a1, const bf16_t* __restrict__ Wt,
              const float* __restrict__ bias, float* __restrict__ out) {
  constexpr int K = 1024;
  int wave = (int)((blockIdx.x * blockDim.x + threadIdx.x) >> 5);
  int lane = (int)(threadIdx.x & 31);
  int row0 = wave * 16;
  int n = lane & 15;
  v8f acc[2];
#pragma unroll
  for (int j = 0; j < 2; ++j) {
    float bv = bias[j * 16 + n];
#pragma unroll
    for (int r = 0; r < 8; ++r) acc[j][r] = bv;
  }
  const bf16_t* wg[2];
#pragma unroll
  for (int j = 0; j < 2; ++j) wg[j] = Wt + (size_t)(j * 16) * K;

  gemm_seg<2>(acc, a1 + (size_t)row0 * K, K, wg, K, 0, K, lane);

  int mrow = row0 + ((lane >> 4) << 3);
#pragma unroll
  for (int j = 0; j < 2; ++j)
#pragma unroll
    for (int r = 0; r < 8; ++r) {
      int row = mrow + r;
      int t = row >> 7;          // row / cB
      int b = row & (cB - 1);    // row % cB
      out[(b * cT + t) * cV + j * 16 + n] = acc[j][r];
    }
}

// ---------------------------------------------------------------------------

extern "C" void kernel_launch(void* const* d_in, const int* in_sizes, int n_in,
                              void* d_out, int out_size, void* d_ws, size_t ws_size,
                              hipStream_t stream) {
  (void)in_sizes; (void)n_in; (void)out_size; (void)ws_size;
  const float SCALE = 16.0f;  // sqrt(E)

  const int*   x   = (const int*)d_in[0];
  const float* emb = (const float*)d_in[1];
  const float* k0f = (const float*)d_in[2];
  const float* r0f = (const float*)d_in[3];
  const float* b0f = (const float*)d_in[4];
  const float* k0b = (const float*)d_in[5];
  const float* r0b = (const float*)d_in[6];
  const float* b0b = (const float*)d_in[7];
  const float* k1f = (const float*)d_in[8];
  const float* r1f = (const float*)d_in[9];
  const float* b1f = (const float*)d_in[10];
  const float* k1b = (const float*)d_in[11];
  const float* r1b = (const float*)d_in[12];
  const float* b1b = (const float*)d_in[13];
  const float* W1  = (const float*)d_in[14];
  const float* d1  = (const float*)d_in[15];
  const float* W2  = (const float*)d_in[16];
  const float* d2  = (const float*)d_in[17];

  // workspace carve-up (256B aligned)
  char* ws = (char*)d_ws;
  size_t off = 0;
  auto take = [&](size_t bytes) -> void* {
    void* p = ws + off;
    off += (bytes + 255) & ~(size_t)255;
    return p;
  };
  bf16_t* x0   = (bf16_t*)take((size_t)cT * cB * cE * 2);
  bf16_t* w0tf = (bf16_t*)take((size_t)4 * cU * (cE + cU) * 2);
  bf16_t* w0tb = (bf16_t*)take((size_t)4 * cU * (cE + cU) * 2);
  bf16_t* w1tf = (bf16_t*)take((size_t)4 * cU * (3 * cU) * 2);
  bf16_t* w1tb = (bf16_t*)take((size_t)4 * cU * (3 * cU) * 2);
  bf16_t* wd1t = (bf16_t*)take((size_t)1024 * 1024 * 2);
  bf16_t* wd2t = (bf16_t*)take((size_t)cV * 1024 * 2);
  bf16_t* hs0  = (bf16_t*)take((size_t)2 * cT * cB * cU * 2);  // fwd ++ bwd
  bf16_t* hs0f = hs0;
  bf16_t* hs0b = hs0 + (size_t)cT * cB * cU;
  bf16_t* hs1f = (bf16_t*)take((size_t)cT * cB * cU * 2);
  bf16_t* hs1b = (bf16_t*)take((size_t)cT * cB * cU * 2);
  float*  c0f  = (float*)take((size_t)cB * cU * 4);
  float*  c0b  = (float*)take((size_t)cB * cU * 4);
  float*  c1f  = (float*)take((size_t)cB * cU * 4);
  float*  c1b  = (float*)take((size_t)cB * cU * 4);
  // dense1 output reuses layer-0 h-sequence (dead after layer-1 scan):
  // T*B*1024 bf16 == 2*T*B*U bf16 exactly.
  bf16_t* out1 = hs0;

  // 1) embedding -> bf16, time-major
  {
    size_t total = (size_t)cT * cB * cE;
    embed_kernel<<<(unsigned)(total / 256), 256, 0, stream>>>(x, emb, x0, SCALE);
  }
  // 2) pack weights (transposed bf16, input||recurrent fused along K)
  {
    size_t tot0 = (size_t)4 * cU * (cE + cU);
    pack_lstm_w<<<(unsigned)((tot0 + 255) / 256), 256, 0, stream>>>(k0f, r0f, w0tf, cE);
    pack_lstm_w<<<(unsigned)((tot0 + 255) / 256), 256, 0, stream>>>(k0b, r0b, w0tb, cE);
    size_t tot1 = (size_t)4 * cU * (3 * cU);
    pack_lstm_w<<<(unsigned)((tot1 + 255) / 256), 256, 0, stream>>>(k1f, r1f, w1tf, 2 * cU);
    pack_lstm_w<<<(unsigned)((tot1 + 255) / 256), 256, 0, stream>>>(k1b, r1b, w1tb, 2 * cU);
    size_t td1 = (size_t)1024 * 1024;
    pack_dense_w<<<(unsigned)((td1 + 255) / 256), 256, 0, stream>>>(W1, wd1t, 1024, 1024);
    size_t td2 = (size_t)1024 * cV;
    pack_dense_w<<<(unsigned)((td2 + 255) / 256), 256, 0, stream>>>(W2, wd2t, 1024, cV);
  }
  // 3) layer-0 bidirectional scan: 512 fused steps (fwd + bwd per launch)
  for (int t = 0; t < cT; ++t) {
    lstm_step<<<64, 256, 0, stream>>>(x0, nullptr, cE, 0, w0tf, w0tb, b0f, b0b,
                                      hs0f, hs0b, c0f, c0b, t);
  }
  // 4) layer-1 bidirectional scan over concat(fwd0, bwd0)
  for (int t = 0; t < cT; ++t) {
    lstm_step<<<64, 256, 0, stream>>>(hs0f, hs0b, cU, cU, w1tf, w1tb, b1f, b1b,
                                      hs1f, hs1b, c1f, c1b, t);
  }
  // 5) dense head
  {
    const int rows = cT * cB;                       // 65536
    const int waves1 = (rows / 16) * 16;            // rowTiles * colGroups
    dense1_kernel<<<(unsigned)(waves1 * 32 / 256), 256, 0, stream>>>(
        hs1f, hs1b, wd1t, d1, out1);
    const int waves2 = rows / 16;                   // 4096
    dense2_kernel<<<(unsigned)(waves2 * 32 / 256), 256, 0, stream>>>(
        out1, wd2t, d2, (float*)d_out);
  }
}
